// ODERNN_10617159155887
// MI455X (gfx1250) — matrix operand order repaired and checked by reading, not compiled, not run on hardware
//
#include <hip/hip_runtime.h>

typedef _Float16 v16h __attribute__((ext_vector_type(16)));
typedef _Float16 v8h  __attribute__((ext_vector_type(8)));
typedef float    v8f  __attribute__((ext_vector_type(8)));

// explicit global-address-space pointers: force global_load_* (not flat_load_*)
// selection inside the noinline body where kernarg addrspace inference can't see.
typedef const _Float16 __attribute__((address_space(1)))* gf16p;
typedef const float    __attribute__((address_space(1)))* gf32p;
typedef const v8h      __attribute__((address_space(1)))* gv8hp;

#define WMMA_F16(a, b, c) \
  __builtin_amdgcn_wmma_f32_16x16x32_f16(false, (a), false, (b), (short)0, (c), false, false)

// ---- problem constants ----
#define BATCH 1024
#define TSEQ  128
#define IDIM  64
#define HDIM  256
#define UDIM  512
#define GDIM  768   // 3*H
#define RK    8

// ---- LDS layout (bytes) ----
#define OFF_ZF   0                    // f32 [16][256]               16384
#define OFF_ACC  16384                // f32 [16][256]   (evolve)    16384
#define OFF_ZB   32768                // f16 [16][264]   (evolve)     8448
#define OFF_H1   41216                // f16 [16][520]   (evolve)    16640
#define OFF_H2   57856                // f16 [16][520]   (evolve)    16640
#define OFF_GI   16384                // f32 [16][768]   (gru)       49152
#define OFF_GH   65536                // f32 [16][768]   (gru)       49152
#define OFF_XB   114688               // f16 [16][72]    (gru)        2304
#define OFF_HB   116992               // f16 [16][264]   (gru)        8448
#define SMEM_BYTES 125440

#define ZB_S 264
#define H1_S 520
#define H2_S 520
#define XB_S 72
#define HB_S 264

static __device__ __forceinline__ v8f vzero8() {
  v8f z = {0.f, 0.f, 0.f, 0.f, 0.f, 0.f, 0.f, 0.f};
  return z;
}

// A fragment (16x32, f16) from LDS: lanes 0-15 rows M=lane, K=0..7 & 16..23; lanes 16-31 K+8
static __device__ __forceinline__ v16h frag_a(const _Float16* base, int stride, int koff) {
  const int lane = threadIdx.x & 31;
  const _Float16* p = base + (lane & 15) * stride + koff + ((lane >> 4) << 3);
  v8h lo = *(const v8h*)(p);
  v8h hi = *(const v8h*)(p + 16);
  v16h v;
#pragma unroll
  for (int i = 0; i < 8; ++i) { v[i] = lo[i]; v[i + 8] = hi[i]; }
  return v;
}

// B fragment (32x16, f16) from GLOBAL: lane = column N; lanes 0-15 K=0..7 & 16..23; lanes 16-31 K+8
// memory layout is N-major / K-fast (= torch (out,in) row-major, no transpose)
static __device__ __forceinline__ v16h frag_b_g(gf16p base, int stride, int koff) {
  const int lane = threadIdx.x & 31;
  gf16p p = base + (lane & 15) * stride + koff + ((lane >> 4) << 3);
  v8h lo = *(gv8hp)(p);
  v8h hi = *(gv8hp)(p + 16);
  v16h v;
#pragma unroll
  for (int i = 0; i < 8; ++i) { v[i] = lo[i]; v[i + 8] = hi[i]; }
  return v;
}

static __device__ __forceinline__ float sigmoidf_(float x) {
  return 1.0f / (1.0f + __expf(-x));
}

// One MLP evaluation (k = func(z_eval)) plus the fused RK4 stage combine.
// noinline is load-bearing: it stops LICM from hoisting the (loop-invariant)
// global weight-fragment loads out of the sstep/kidx loops, which previously
// ballooned past 256 VGPRs and spilled every B fragment to scratch memory.
__device__ __attribute__((noinline)) void func_eval(
    const _Float16* __restrict__ wb1_, const float* __restrict__ fb1_,
    const _Float16* __restrict__ wb2_, const float* __restrict__ fb2_,
    const _Float16* __restrict__ wb3_, const float* __restrict__ fb3_,
    int kidx, float dt) {
  extern __shared__ char smem[];
  float*    zf  = (float*)(smem + OFF_ZF);
  float*    acc = (float*)(smem + OFF_ACC);
  _Float16* zb  = (_Float16*)(smem + OFF_ZB);
  _Float16* h1b = (_Float16*)(smem + OFF_H1);
  _Float16* h2b = (_Float16*)(smem + OFF_H2);

  gf16p wb1 = (gf16p)wb1_;
  gf16p wb2 = (gf16p)wb2_;
  gf16p wb3 = (gf16p)wb3_;
  gf32p fb1 = (gf32p)fb1_;
  gf32p fb2 = (gf32p)fb2_;
  gf32p fb3 = (gf32p)fb3_;

  const int wave = threadIdx.x >> 5;
  const int lane = threadIdx.x & 31;

  // ---- L1: h1 = tanh(z @ W1^T + b1)   [16x512, K=256] ----
  {
    v8f c4[4] = {vzero8(), vzero8(), vzero8(), vzero8()};
    const int nb = wave * 64;
#pragma unroll
    for (int kc = 0; kc < 8; ++kc) {
      v16h a = frag_a(zb, ZB_S, kc * 32);
#pragma unroll
      for (int j = 0; j < 4; ++j) {
        v16h b = frag_b_g(wb1 + (nb + j * 16) * HDIM, HDIM, kc * 32);
        c4[j] = WMMA_F16(a, b, c4[j]);
      }
    }
    const int rb = (lane >> 4) << 3;
#pragma unroll
    for (int j = 0; j < 4; ++j) {
      const int col  = nb + j * 16 + (lane & 15);
      const float bs = fb1[col];
#pragma unroll
      for (int i = 0; i < 8; ++i)
        h1b[(rb + i) * H1_S + col] = (_Float16)tanhf(c4[j][i] + bs);
    }
  }
  __syncthreads();
  // ---- L2: h2 = tanh(h1 @ W2^T + b2)  [16x512, K=512] ----
  {
    v8f c4[4] = {vzero8(), vzero8(), vzero8(), vzero8()};
    const int nb = wave * 64;
#pragma unroll
    for (int kc = 0; kc < 16; ++kc) {
      v16h a = frag_a(h1b, H1_S, kc * 32);
#pragma unroll
      for (int j = 0; j < 4; ++j) {
        v16h b = frag_b_g(wb2 + (nb + j * 16) * UDIM, UDIM, kc * 32);
        c4[j] = WMMA_F16(a, b, c4[j]);
      }
    }
    const int rb = (lane >> 4) << 3;
#pragma unroll
    for (int j = 0; j < 4; ++j) {
      const int col  = nb + j * 16 + (lane & 15);
      const float bs = fb2[col];
#pragma unroll
      for (int i = 0; i < 8; ++i)
        h2b[(rb + i) * H2_S + col] = (_Float16)tanhf(c4[j][i] + bs);
    }
  }
  __syncthreads();
  // ---- L3: k = h2 @ W3^T + b3  [16x256, K=512] + fused RK4 combine ----
  {
    v8f d2[2] = {vzero8(), vzero8()};
    const int nb = wave * 32;
#pragma unroll
    for (int kc = 0; kc < 16; ++kc) {
      v16h a = frag_a(h2b, H2_S, kc * 32);
#pragma unroll
      for (int j = 0; j < 2; ++j) {
        v16h b = frag_b_g(wb3 + (nb + j * 16) * UDIM, UDIM, kc * 32);
        d2[j] = WMMA_F16(a, b, d2[j]);
      }
    }
    const int rb = (lane >> 4) << 3;
#pragma unroll
    for (int j = 0; j < 2; ++j) {
      const int col  = nb + j * 16 + (lane & 15);
      const float bs = fb3[col];
#pragma unroll
      for (int i = 0; i < 8; ++i) {
        const int row = rb + i;
        const int idx = row * HDIM + col;
        const float kv = d2[j][i] + bs;
        const float zc = zf[idx];
        if (kidx == 0) {
          acc[idx] = kv;
          zb[row * ZB_S + col] = (_Float16)(zc + 0.5f * dt * kv);
        } else if (kidx == 1) {
          acc[idx] += 2.0f * kv;
          zb[row * ZB_S + col] = (_Float16)(zc + 0.5f * dt * kv);
        } else if (kidx == 2) {
          acc[idx] += 2.0f * kv;
          zb[row * ZB_S + col] = (_Float16)(zc + dt * kv);
        } else {
          zf[idx] = zc + (dt / 6.0f) * (acc[idx] + kv);
        }
      }
    }
  }
  __syncthreads();
}

__global__ void __launch_bounds__(256)
ode_step_kernel(const float* __restrict__ x, const float* __restrict__ times, int t,
                const _Float16* __restrict__ wb_ih_, const _Float16* __restrict__ wb_hh_,
                const float* __restrict__ b_ih, const float* __restrict__ b_hh,
                const _Float16* __restrict__ wb1, const float* __restrict__ fb1,
                const _Float16* __restrict__ wb2, const float* __restrict__ fb2,
                const _Float16* __restrict__ wb3, const float* __restrict__ fb3,
                float* __restrict__ h_state) {
  extern __shared__ char smem[];
  float*    zf  = (float*)(smem + OFF_ZF);
  _Float16* zb  = (_Float16*)(smem + OFF_ZB);
  float*    gib = (float*)(smem + OFF_GI);
  float*    ghb = (float*)(smem + OFF_GH);
  _Float16* xb  = (_Float16*)(smem + OFF_XB);
  _Float16* hb  = (_Float16*)(smem + OFF_HB);

  gf16p wb_ih = (gf16p)wb_ih_;
  gf16p wb_hh = (gf16p)wb_hh_;

  const int tid  = threadIdx.x;
  const int wave = tid >> 5;
  const int lane = tid & 31;
  const int m0   = blockIdx.x * 16;

  // load this block's 16 rows of h
  for (int i = tid; i < 16 * HDIM; i += 256) zf[i] = h_state[m0 * HDIM + i];
  __syncthreads();

  // ---------------- evolve: 8 fixed RK4 steps of dz/dt = MLP(z) ----------------
  if (t > 0) {
    const float dtt = times[t] - times[t - 1];
    const float dt  = dtt / (float)RK;
#pragma unroll 1
    for (int sstep = 0; sstep < RK; ++sstep) {
      for (int i = tid; i < 16 * HDIM; i += 256) {
        int r = i >> 8, c = i & 255;
        zb[r * ZB_S + c] = (_Float16)zf[i];
      }
      __syncthreads();
#pragma unroll 1
      for (int kidx = 0; kidx < 4; ++kidx) {
        func_eval(wb1, fb1, wb2, fb2, wb3, fb3, kidx, dt);
      }
    }
  }

  // ---------------- GRU cell: h = GRU(x_t, h) ----------------
  for (int i = tid; i < 16 * IDIM; i += 256) {
    int r = i >> 6, c = i & 63;
    xb[r * XB_S + c] = (_Float16)x[(m0 + r) * (TSEQ * IDIM) + t * IDIM + c];
  }
  for (int i = tid; i < 16 * HDIM; i += 256) {
    int r = i >> 8, c = i & 255;
    hb[r * HB_S + c] = (_Float16)zf[i];
  }
  __syncthreads();
  {
    const int nb = wave * 96;  // 6 tiles of 16 over N=768
    // gh = h @ W_hh^T   [16x768, K=256]
    {
      v8f g[6] = {vzero8(), vzero8(), vzero8(), vzero8(), vzero8(), vzero8()};
#pragma unroll
      for (int kc = 0; kc < 8; ++kc) {
        v16h a = frag_a(hb, HB_S, kc * 32);
#pragma unroll
        for (int j = 0; j < 6; ++j) {
          v16h b = frag_b_g(wb_hh + (nb + j * 16) * HDIM, HDIM, kc * 32);
          g[j] = WMMA_F16(a, b, g[j]);
        }
      }
      const int rb = (lane >> 4) << 3;
#pragma unroll
      for (int j = 0; j < 6; ++j) {
        const int col = nb + j * 16 + (lane & 15);
#pragma unroll
        for (int i = 0; i < 8; ++i) ghb[(rb + i) * GDIM + col] = g[j][i];
      }
    }
    // gi = x_t @ W_ih^T  [16x768, K=64]
    {
      v8f g[6] = {vzero8(), vzero8(), vzero8(), vzero8(), vzero8(), vzero8()};
#pragma unroll
      for (int kc = 0; kc < 2; ++kc) {
        v16h a = frag_a(xb, XB_S, kc * 32);
#pragma unroll
        for (int j = 0; j < 6; ++j) {
          v16h b = frag_b_g(wb_ih + (nb + j * 16) * IDIM, IDIM, kc * 32);
          g[j] = WMMA_F16(a, b, g[j]);
        }
      }
      const int rb = (lane >> 4) << 3;
#pragma unroll
      for (int j = 0; j < 6; ++j) {
        const int col = nb + j * 16 + (lane & 15);
#pragma unroll
        for (int i = 0; i < 8; ++i) gib[(rb + i) * GDIM + col] = g[j][i];
      }
    }
  }
  __syncthreads();
  for (int i = tid; i < 16 * HDIM; i += 256) {
    int r = i >> 8, c = i & 255;
    float ir = gib[r * GDIM + c] + b_ih[c];
    float hr = ghb[r * GDIM + c] + b_hh[c];
    float iz = gib[r * GDIM + c + 256] + b_ih[c + 256];
    float hz = ghb[r * GDIM + c + 256] + b_hh[c + 256];
    float in_ = gib[r * GDIM + c + 512] + b_ih[c + 512];
    float hn = ghb[r * GDIM + c + 512] + b_hh[c + 512];
    float rg = sigmoidf_(ir + hr);
    float zg = sigmoidf_(iz + hz);
    float ng = tanhf(in_ + rg * hn);
    h_state[m0 * HDIM + i] = (1.0f - zg) * ng + zg * zf[i];
  }
}

__global__ void cvt_f32_to_f16(const float* __restrict__ s, _Float16* __restrict__ d, int n) {
  int i = blockIdx.x * blockDim.x + threadIdx.x;
  if (i < n) d[i] = (_Float16)s[i];
}

__global__ void zero_f32(float* __restrict__ p, int n) {
  int i = blockIdx.x * blockDim.x + threadIdx.x;
  if (i < n) p[i] = 0.0f;
}

// out = h @ w_out^T + b_out  (1024x20, 10 MFLOP -> plain f32)
__global__ void out_proj(const float* __restrict__ h, const float* __restrict__ w,
                         const float* __restrict__ b, float* __restrict__ out) {
  int i = blockIdx.x * blockDim.x + threadIdx.x;
  if (i >= BATCH * 20) return;
  int bi = i / 20, o = i % 20;
  float s = b[o];
  for (int k = 0; k < HDIM; ++k) s += h[bi * HDIM + k] * w[o * HDIM + k];
  out[i] = s;
}

// ---- workspace layout (bytes) ----
#define WS_WIH 0u          // f16 768*64
#define WS_WHH 98304u      // f16 768*256
#define WS_W1  491520u     // f16 512*256
#define WS_W2  753664u     // f16 512*512
#define WS_W3  1277952u    // f16 256*512
#define WS_H   1540096u    // f32 1024*256

extern "C" void kernel_launch(void* const* d_in, const int* in_sizes, int n_in,
                              void* d_out, int out_size, void* d_ws, size_t ws_size,
                              hipStream_t stream) {
  const float* x     = (const float*)d_in[0];
  const float* times = (const float*)d_in[1];
  const float* b_ih  = (const float*)d_in[4];
  const float* b_hh  = (const float*)d_in[5];
  const float* fb1   = (const float*)d_in[7];
  const float* fb2   = (const float*)d_in[9];
  const float* fb3   = (const float*)d_in[11];
  const float* w_out = (const float*)d_in[12];
  const float* b_out = (const float*)d_in[13];

  char* ws = (char*)d_ws;
  _Float16* wb_ih = (_Float16*)(ws + WS_WIH);
  _Float16* wb_hh = (_Float16*)(ws + WS_WHH);
  _Float16* wb1   = (_Float16*)(ws + WS_W1);
  _Float16* wb2   = (_Float16*)(ws + WS_W2);
  _Float16* wb3   = (_Float16*)(ws + WS_W3);
  float*    hst   = (float*)(ws + WS_H);

  auto cvt = [&](const void* s, _Float16* d, int n) {
    cvt_f32_to_f16<<<(n + 255) / 256, 256, 0, stream>>>((const float*)s, d, n);
  };
  cvt(d_in[2], wb_ih, GDIM * IDIM);
  cvt(d_in[3], wb_hh, GDIM * HDIM);
  cvt(d_in[6], wb1, UDIM * HDIM);
  cvt(d_in[8], wb2, UDIM * UDIM);
  cvt(d_in[10], wb3, HDIM * UDIM);
  zero_f32<<<(BATCH * HDIM + 255) / 256, 256, 0, stream>>>(hst, BATCH * HDIM);

  for (int t = 0; t < TSEQ; ++t) {
    ode_step_kernel<<<BATCH / 16, 256, SMEM_BYTES, stream>>>(
        x, times, t, wb_ih, wb_hh, b_ih, b_hh, wb1, fb1, wb2, fb2, wb3, fb3, hst);
  }
  out_proj<<<(BATCH * 20 + 255) / 256, 256, 0, stream>>>(hst, w_out, b_out, (float*)d_out);
}
